// Beta_cVAE_18116172055026
// MI455X (gfx1250) — matrix operand order repaired and checked by reading, not compiled, not run on hardware
//
#include <hip/hip_runtime.h>

// ---------------------------------------------------------------------------
// Beta-cVAE + batched trajectory-projection solver for MI455X (gfx1250).
//  - MLP layers use v_wmma_f32_16x16x32_f16 (f16 in, f32 accumulate),
//    register-blocked 32x64 per wave: 8 WMMAs per 6 fragment loads.
//  - Solver: one wave32 per batch row; atan2/cos/sin folded into rsqrt;
//    big (B,1000)x(1000,11) products reduced to Gram-matrix algebra.
// ---------------------------------------------------------------------------

#define NUMT 100
#define NV   11
#define NOBS 10
#define A_OBS_C 8.0f
#define B_OBS_C 4.2f
#define KD_C 8.944271909999159f   // 2*sqrt(20)
#define V_MIN_C 0.1f
#define V_MAX_C 30.0f
#define A_MAX_C 8.0f
#define RHO_OBS_C 100.0f
#define RHO_INEQ_C 100.0f
#define RHO_LANE_C 100.0f
#define RHO_PROJ_C 1.0f
#define MAXIT 20

// constant-pack offsets (floats) inside workspace
#define OFF_PTP      0      // P^T P            (121)
#define OFF_PDTPD    121    // Pdot^T Pdot      (121)
#define OFF_PDDTPDD  242    // Pddot^T Pddot    (121)
#define OFF_INV1X    363    // inv1_x 14x14     (196)
#define OFF_INV1Y    559    // inv1_y 15x15     (225)
#define OFF_INV2X    784    // inv2_x 14x14     (196)
#define OFF_INV2Y    980    // inv2_y 15x15     (225)
#define OFF_SAVD     1205   // block sums of A_vd (4x11)
#define OFF_SAPD     1249   // block sums of A_pd (4x11)
#define OFF_SP       1293   // sum_t P[t,k]       (11)
#define OFF_SPT      1304   // sum_t t_time*P     (11)
#define CPACK_N      1315

typedef __attribute__((ext_vector_type(16))) _Float16 v16h;
typedef __attribute__((ext_vector_type(8)))  float    v8f;

__device__ __forceinline__ float wave_sum(float v) {
#pragma unroll
  for (int off = 16; off > 0; off >>= 1) v += __shfl_xor(v, off, 32);
  return v;
}

// ---------------------------------------------------------------------------
// Gauss-Jordan inverse (no pivoting; valid for these KKT matrices: SPD 11x11
// leading block -> first 11 pivots > 0, Schur complement ND -> pivots < 0).
// a is n x 2n augmented [M | I]; on exit right half holds M^-1.
// ---------------------------------------------------------------------------
__device__ void gj_invert(float* a, int n) {
  const int w = 2 * n;
  for (int c = 0; c < n; ++c) {
    float ip = 1.0f / a[c * w + c];
    for (int j = 0; j < w; ++j) a[c * w + j] *= ip;
    for (int r = 0; r < n; ++r) {
      if (r == c) continue;
      float f = a[r * w + c];
      for (int j = 0; j < w; ++j) a[r * w + j] -= f * a[c * w + j];
    }
  }
}

// ---------------------------------------------------------------------------
// One-block setup: Gram matrices, lincost block-sums, obstacle projection
// vectors, and the four KKT inverses.
// ---------------------------------------------------------------------------
__global__ void __launch_bounds__(256) prep_const_kernel(
    const float* __restrict__ P, const float* __restrict__ Pd,
    const float* __restrict__ Pdd, float* __restrict__ cpack)
{
  __shared__ float sPtP[121], sPdTPd[121], sPddTPdd[121], sAvT[121], sApT[121];
  __shared__ float sGJ[4][15 * 30];
  const int tid = threadIdx.x;

  // 5 Gram matrices: P'P, Pd'Pd, Pdd'Pdd, Avd'Avd, Apd'Apd
  for (int q = tid; q < 5 * 121; q += 256) {
    int m = q / 121, ij = q % 121, i = ij / 11, j = ij % 11;
    float s = 0.f;
    for (int t = 0; t < NUMT; ++t) {
      float ai, aj;
      if (m == 0)      { ai = P[t*11+i];   aj = P[t*11+j]; }
      else if (m == 1) { ai = Pd[t*11+i];  aj = Pd[t*11+j]; }
      else if (m == 2) { ai = Pdd[t*11+i]; aj = Pdd[t*11+j]; }
      else if (m == 3) { ai = Pdd[t*11+i] - 20.f*Pd[t*11+i];
                         aj = Pdd[t*11+j] - 20.f*Pd[t*11+j]; }
      else             { ai = Pdd[t*11+i] - 20.f*P[t*11+i] - KD_C*Pd[t*11+i];
                         aj = Pdd[t*11+j] - 20.f*P[t*11+j] - KD_C*Pd[t*11+j]; }
      s += ai * aj;
    }
    float* dst = (m==0)?sPtP:(m==1)?sPdTPd:(m==2)?sPddTPdd:(m==3)?sAvT:sApT;
    dst[ij] = s;
  }
  // block sums for v_des/y_des lincosts (jnp.repeat -> 4 blocks of 25)
  for (int q = tid; q < 44; q += 256) {
    int blk = q / 11, k = q % 11;
    float sv = 0.f, sp = 0.f;
    for (int t = blk * 25; t < blk * 25 + 25; ++t) {
      sv += Pdd[t*11+k] - 20.f*Pd[t*11+k];
      sp += Pdd[t*11+k] - 20.f*P[t*11+k] - KD_C*Pd[t*11+k];
    }
    cpack[OFF_SAVD + q] = sv;
    cpack[OFF_SAPD + q] = sp;
  }
  // SP[k]=sum_t P, SPt[k]=sum_t t_time*P  (obstacle trajectory projections)
  for (int k = tid; k < 11; k += 256) {
    float s0 = 0.f, s1 = 0.f;
    for (int t = 0; t < NUMT; ++t) {
      s0 += P[t*11+k];
      s1 += ((float)t * (15.0f / 99.0f)) * P[t*11+k];
    }
    cpack[OFF_SP + k] = s0;
    cpack[OFF_SPT + k] = s1;
  }
  __syncthreads();
  for (int q = tid; q < 121; q += 256) {
    cpack[OFF_PTP + q]     = sPtP[q];
    cpack[OFF_PDTPD + q]   = sPdTPd[q];
    cpack[OFF_PDDTPDD + q] = sPddTPdd[q];
  }
  // build 4 augmented KKT matrices: m0=M1x(14) m1=M1y(15) m2=M2x(14) m3=M2y(15)
  for (int m = 0; m < 4; ++m) {
    const int n = (m & 1) ? 15 : 14;
    const int w2 = 2 * n;
    for (int q = tid; q < n * n; q += 256) {
      int i = q / n, j = q % n;
      float v;
      if (i < 11 && j < 11) {
        if (m == 0)      v = sPddTPdd[i*11+j] + sAvT[i*11+j];                  // cost1_x
        else if (m == 1) v = sPddTPdd[i*11+j] + sApT[i*11+j];                  // cost1_y
        else v = ((i == j) ? 1.f : 0.f)
               + ((m == 2) ? 1000.f : 1200.f) * sPtP[i*11+j]                    // rho_obs*10 (+rho_lane*2)
               + 100.f * (sPddTPdd[i*11+j] + sPdTPd[i*11+j]);                   // rho_ineq*(acc+vel)
      } else if (i >= 11 && j >= 11) {
        v = 0.f;
      } else {
        int r = (i >= 11) ? (i - 11) : (j - 11);
        int c = (i >= 11) ? j : i;
        v = (r == 0) ? P[c] : (r == 1) ? Pd[c] : (r == 2) ? Pdd[c] : Pd[99*11 + c];
      }
      sGJ[m][i * w2 + j]     = v;
      sGJ[m][i * w2 + n + j] = (i == j) ? 1.f : 0.f;
    }
  }
  __syncthreads();
  if (tid < 4) gj_invert(sGJ[tid], (tid & 1) ? 15 : 14);
  __syncthreads();
  for (int q = tid; q < 196; q += 256) {
    int i = q / 14, j = q % 14;
    cpack[OFF_INV1X + q] = sGJ[0][i * 28 + 14 + j];
    cpack[OFF_INV2X + q] = sGJ[2][i * 28 + 14 + j];
  }
  for (int q = tid; q < 225; q += 256) {
    int i = q / 15, j = q % 15;
    cpack[OFF_INV1Y + q] = sGJ[1][i * 30 + 15 + j];
    cpack[OFF_INV2Y + q] = sGJ[3][i * 30 + 15 + j];
  }
}

// ---------------------------------------------------------------------------
// Weight transpose + f32->f16 with K padding:  Wt[n*Kpad + k] = W[k*N + n]
// ---------------------------------------------------------------------------
__global__ void transpose_pad_f16(const float* __restrict__ W,
                                  _Float16* __restrict__ Wt,
                                  int K, int N, int Kpad)
{
  const int total = N * Kpad;
  for (int idx = blockIdx.x * blockDim.x + threadIdx.x; idx < total;
       idx += gridDim.x * blockDim.x) {
    int n = idx / Kpad, k = idx % Kpad;
    Wt[idx] = (k < K) ? (_Float16)W[(size_t)k * N + n] : (_Float16)0.f;
  }
}

// encoder input: [inp_n(55) | traj_gt(200) | 0] padded to 256, f16
__global__ void build_x1_kernel(const float* __restrict__ inp,
                                const float* __restrict__ traj,
                                const float* __restrict__ mean,
                                const float* __restrict__ stdv,
                                _Float16* __restrict__ x1,
                                float* __restrict__ inpn)
{
  for (int idx = blockIdx.x * blockDim.x + threadIdx.x; idx < 1024 * 256;
       idx += gridDim.x * blockDim.x) {
    int b = idx >> 8, c = idx & 255;
    float v;
    if (c < 55)       { v = (inp[b*55 + c] - mean[c]) / stdv[c]; inpn[b*55 + c] = v; }
    else if (c < 255) { v = traj[b*200 + (c - 55)]; }
    else              { v = 0.f; }
    x1[idx] = (_Float16)v;
  }
}

// ---------------------------------------------------------------------------
// WMMA GEMM:  C(MxN,f16) = act(A(MxKpad,f16) @ Wt^T + bias), Wt is N x Kpad.
// Wave computes a 32x64 macro-tile = 2x4 accumulators: per K-step, 2 A-frags
// + 4 B-frags feed 8 v_wmma_f32_16x16x32_f16 (2.7x wmma-per-load vs 1 tile).
// Fragment packing per CDNA5 ISA 7.12.2 VGPR tables. Requires M%32==0, N%64==0.
// ---------------------------------------------------------------------------
__global__ void __launch_bounds__(256) wmma_gemm_relu(
    const _Float16* __restrict__ A, const _Float16* __restrict__ Bt,
    const float* __restrict__ bias, _Float16* __restrict__ C,
    int M, int N, int K, int relu)
{
  const int wave = blockIdx.x * (blockDim.x >> 5) + (threadIdx.x >> 5);
  const int lane = threadIdx.x & 31;
  const int tilesN = N >> 6;            // 64-wide N macro-columns
  const int m0 = (wave / tilesN) << 5;  // 32-tall M macro-rows
  const int n0 = (wave % tilesN) << 6;
  if (m0 >= M) return;

  const int mr = lane & 15;
  const int hi = lane >> 4;            // lane half
  const int koffA = hi ? 8 : 0;        // A: K-split 0-7 / 8-15 within 16-block
  const int koffB = hi ? 8 : 0;        // B: same K-split as A

  const _Float16* Arow[2];
  const _Float16* Brow[4];
#pragma unroll
  for (int mi = 0; mi < 2; ++mi) Arow[mi] = A  + (size_t)(m0 + 16 * mi + mr) * K;
#pragma unroll
  for (int ni = 0; ni < 4; ++ni) Brow[ni] = Bt + (size_t)(n0 + 16 * ni + mr) * K;

  v8f acc[2][4] = {};
  for (int k0 = 0; k0 < K; k0 += 32) {
    union { v16h v; unsigned u[8]; } af[2], bf[4];
#pragma unroll
    for (int g = 0; g < 8; ++g) {
      const int ka = ((g < 4) ? (g << 1) : (16 + ((g - 4) << 1))) + koffA;
      const int kb = ((g < 4) ? (g << 1) : (16 + ((g - 4) << 1))) + koffB;
#pragma unroll
      for (int mi = 0; mi < 2; ++mi) af[mi].u[g] = *(const unsigned*)(Arow[mi] + k0 + ka);
#pragma unroll
      for (int ni = 0; ni < 4; ++ni) bf[ni].u[g] = *(const unsigned*)(Brow[ni] + k0 + kb);
    }
    if (k0 + 32 < K) {                         // gfx1250 global_prefetch_b8
      __builtin_prefetch(Arow[0] + k0 + 32, 0, 1);
      __builtin_prefetch(Arow[1] + k0 + 32, 0, 1);
      __builtin_prefetch(Brow[0] + k0 + 32, 0, 1);
      __builtin_prefetch(Brow[2] + k0 + 32, 0, 1);
    }
#pragma unroll
    for (int mi = 0; mi < 2; ++mi)
#pragma unroll
      for (int ni = 0; ni < 4; ++ni)
        acc[mi][ni] = __builtin_amdgcn_wmma_f32_16x16x32_f16(
            false, af[mi].v, false, bf[ni].v, (short)0, acc[mi][ni], false, false);
  }

  // C/D layout: VGPR r -> M = mbase + r (+8 for upper lanes), N = nbase + lane%16
#pragma unroll
  for (int ni = 0; ni < 4; ++ni) {
    const int n = n0 + 16 * ni + (lane & 15);
    const float bz = bias ? bias[n] : 0.f;
#pragma unroll
    for (int mi = 0; mi < 2; ++mi) {
      const int mb = m0 + 16 * mi + (hi ? 8 : 0);
#pragma unroll
      for (int r = 0; r < 8; ++r) {
        float v = acc[mi][ni][r] + bz;
        if (relu) v = fmaxf(v, 0.f);
        C[(size_t)(mb + r) * N + n] = (_Float16)v;
      }
    }
  }
}

// mu/logvar heads (N=2) + reparameterization: one block per row, wave per dot
__global__ void __launch_bounds__(128) muvarz_kernel(
    const _Float16* __restrict__ h2, const float* __restrict__ wmu,
    const float* __restrict__ bmu, const float* __restrict__ wlv,
    const float* __restrict__ blv, const float* __restrict__ eps,
    float* __restrict__ z)
{
  const int b = blockIdx.x;
  const int wave = threadIdx.x >> 5, lane = threadIdx.x & 31;
  const int j = wave & 1;
  const float* W = (wave >= 2) ? wlv : wmu;
  float s = 0.f;
  for (int k = lane; k < 1024; k += 32)
    s += (float)h2[(size_t)b * 1024 + k] * W[k * 2 + j];
  s = wave_sum(s);
  __shared__ float sm[4];
  if (lane == 0) sm[wave] = s;
  __syncthreads();
  if (threadIdx.x < 2) {
    int jj = threadIdx.x;
    float mu = sm[jj] + bmu[jj];
    float lv = sm[2 + jj] + blv[jj];
    z[b * 2 + jj] = mu + __expf(0.5f * lv) * eps[b * 2 + jj];
  }
}

// decoder input: [z(2) | inp_n(55) | 0] padded to 64, f16
__global__ void build_x2_kernel(const float* __restrict__ z,
                                const float* __restrict__ inpn,
                                _Float16* __restrict__ x2)
{
  for (int idx = blockIdx.x * blockDim.x + threadIdx.x; idx < 1024 * 64;
       idx += gridDim.x * blockDim.x) {
    int b = idx >> 6, c = idx & 63;
    float v = (c < 2) ? z[b * 2 + c] : (c < 57) ? inpn[b * 55 + (c - 2)] : 0.f;
    x2[idx] = (_Float16)v;
  }
}

// final decoder layer (N=8): block per row, wave per output column
__global__ void __launch_bounds__(256) dec3_kernel(
    const _Float16* __restrict__ h4, const float* __restrict__ w3,
    const float* __restrict__ b3, float* __restrict__ nn_out)
{
  const int b = blockIdx.x;
  const int wave = threadIdx.x >> 5, lane = threadIdx.x & 31;
  float s = 0.f;
  for (int k = lane; k < 1024; k += 32)
    s += (float)h4[(size_t)b * 1024 + k] * w3[k * 8 + wave];
  s = wave_sum(s);
  if (lane == 0) nn_out[b * 8 + wave] = s + b3[wave];
}

// ---------------------------------------------------------------------------
// Projection solver: one wave32 per batch row, 20 iterations, lanes over t.
// cos/sin(atan2) -> rsqrt; matvecs through precomputed Gram matrices.
// ---------------------------------------------------------------------------
__global__ void __launch_bounds__(128) solver_kernel(
    const float* __restrict__ inp, const float* __restrict__ ego,
    const float* __restrict__ yub_g, const float* __restrict__ ylb_g,
    const float* __restrict__ nn_out, const float* __restrict__ Pg,
    const float* __restrict__ Pdg, const float* __restrict__ Pddg,
    const float* __restrict__ cpack, float* __restrict__ out)
{
  __shared__ float sP[NUMT * NV], sPd[NUMT * NV], sPdd[NUMT * NV];
  __shared__ float sPtP[121], sPdTPd[121], sPddTPdd[121];
  __shared__ float sInv1x[196], sInv2x[196], sInv1y[225], sInv2y[225];
  __shared__ float sSAvd[44], sSApd[44], sSP[11], sSPt[11];
  __shared__ float sObs[4][40];

  const int tid = threadIdx.x;
  for (int q = tid; q < NUMT * NV; q += 128) { sP[q] = Pg[q]; sPd[q] = Pdg[q]; sPdd[q] = Pddg[q]; }
  for (int q = tid; q < 121; q += 128) {
    sPtP[q] = cpack[OFF_PTP + q];
    sPdTPd[q] = cpack[OFF_PDTPD + q];
    sPddTPdd[q] = cpack[OFF_PDDTPDD + q];
  }
  for (int q = tid; q < 196; q += 128) { sInv1x[q] = cpack[OFF_INV1X + q]; sInv2x[q] = cpack[OFF_INV2X + q]; }
  for (int q = tid; q < 225; q += 128) { sInv1y[q] = cpack[OFF_INV1Y + q]; sInv2y[q] = cpack[OFF_INV2Y + q]; }
  for (int q = tid; q < 44; q += 128)  { sSAvd[q] = cpack[OFF_SAVD + q]; sSApd[q] = cpack[OFF_SAPD + q]; }
  if (tid < 11) { sSP[tid] = cpack[OFF_SP + tid]; sSPt[tid] = cpack[OFF_SPT + tid]; }

  const int wave = tid >> 5, lane = tid & 31;
  const int b = blockIdx.x * 4 + wave;
  if (lane < NOBS) {  // inp[:,5::5] etc.
    sObs[wave][lane]      = inp[b * 55 + 5 + 5 * lane];
    sObs[wave][10 + lane] = inp[b * 55 + 6 + 5 * lane];
    sObs[wave][20 + lane] = inp[b * 55 + 7 + 5 * lane];
    sObs[wave][30 + lane] = inp[b * 55 + 8 + 5 * lane];
  }
  __syncthreads();

  const float beqx1 = ego[b * 4 + 2], beqy1 = ego[b * 4 + 3];
  const float yub = yub_g[b], ylb = ylb_g[b];
  float nn0[8];
#pragma unroll
  for (int j = 0; j < 8; ++j) nn0[j] = nn_out[b * 8 + j];

  float cxb[NV], cyb[NV], cx[NV], cy[NV], lx[NV], ly[NV], xoP[NV], yoP[NV];
  {
    float lcx[NV], lcy[NV];
#pragma unroll
    for (int k = 0; k < NV; ++k) {
      float sx = 0.f, sy = 0.f;
#pragma unroll
      for (int j = 0; j < 4; ++j) { sx += nn0[j] * sSAvd[j * 11 + k]; sy += nn0[4 + j] * sSApd[j * 11 + k]; }
      lcx[k] = 20.f * sx;   // lincost_x1 = K_P_V * v_des @ A_vd
      lcy[k] = 20.f * sy;   // lincost_y1 = K_P   * y_des @ A_pd
    }
#pragma unroll
    for (int k = 0; k < NV; ++k) {
      float ax = sInv1x[k * 14 + 12] * beqx1;  // b_eq = [0, v0, 0(,0)]
      float ay = sInv1y[k * 15 + 12] * beqy1;
#pragma unroll
      for (int j = 0; j < NV; ++j) { ax -= sInv1x[k * 14 + j] * lcx[j]; ay -= sInv1y[k * 15 + j] * lcy[j]; }
      cxb[k] = ax; cyb[k] = ay; cx[k] = ax; cy[k] = ay; lx[k] = 0.f; ly[k] = 0.f;
    }
    float sxo = 0.f, syo = 0.f, svx = 0.f, svy = 0.f;
    for (int o = 0; o < NOBS; ++o) {
      sxo += sObs[wave][o]; syo += sObs[wave][10 + o];
      svx += sObs[wave][20 + o]; svy += sObs[wave][30 + o];
    }
#pragma unroll
    for (int k = 0; k < NV; ++k) {  // x_obs_traj @ A_obs_m (iteration-invariant)
      xoP[k] = sxo * sSP[k] + svx * sSPt[k];
      yoP[k] = syo * sSP[k] + svy * sSPt[k];
    }
  }

  for (int it = 0; it < MAXIT; ++it) {
    float aUX[NV] = {}, aUY[NV] = {}, aAX[NV] = {}, aAY[NV] = {};
    float aVX[NV] = {}, aVY[NV] = {}, aL1[NV] = {}, aL2[NV] = {};
    for (int t = lane; t < NUMT; t += 32) {
      float x = 0.f, y = 0.f, xd = 0.f, yd = 0.f, xdd = 0.f, ydd = 0.f;
#pragma unroll
      for (int j = 0; j < NV; ++j) {
        float pj = sP[t * NV + j], pdj = sPd[t * NV + j], pddj = sPdd[t * NV + j];
        x += cx[j] * pj;  y += cy[j] * pj;
        xd += cx[j] * pdj; yd += cy[j] * pdj;
        xdd += cx[j] * pddj; ydd += cy[j] * pddj;
      }
      // velocity projection: cos/sin(atan2(yd,xd)) via rsqrt; dot == |v|
      float cv, sv, dv;
      float rv2 = xd * xd + yd * yd;
      if (rv2 < 1e-24f) { cv = 1.f; sv = 0.f; dv = V_MIN_C; }
      else { float ir = rsqrtf(rv2); cv = xd * ir; sv = yd * ir;
             dv = fminf(fmaxf(rv2 * ir, V_MIN_C), V_MAX_C); }
      float dvcv = dv * cv, dvsv = dv * sv;
      // acceleration projection
      float caa, saa, da;
      float ra2 = xdd * xdd + ydd * ydd;
      if (ra2 < 1e-24f) { caa = 1.f; saa = 0.f; da = 0.f; }
      else { float ir = rsqrtf(ra2); caa = xdd * ir; saa = ydd * ir;
             da = fminf(ra2 * ir, A_MAX_C); }
      float daca = da * caa, dasa = da * saa;
      // lane terms: res_lane = relu(y-yub)-relu(ylb-y); (b-s) = min(yub,y)+max(ylb,y)
      float l1 = fmaxf(y - yub, 0.f) - fmaxf(ylb - y, 0.f);
      float l2 = fminf(yub, y) + fmaxf(ylb, y);
      float tt = (float)t * (15.0f / 99.0f);
      float sux = 0.f, suy = 0.f;   // sum over obstacles of A*d_o*ca, B*d_o*sa
      for (int o = 0; o < NOBS; ++o) {
        float xo = sObs[wave][o] + sObs[wave][20 + o] * tt;
        float yo = sObs[wave][10 + o] + sObs[wave][30 + o] * tt;
        float wc = x - xo, ws = y - yo;
        float aws = A_OBS_C * ws, bwc = B_OBS_C * wc;
        float r2 = aws * aws + bwc * bwc;
        float ca, sa;
        if (r2 < 1e-24f) { ca = 1.f; sa = 0.f; }
        else { float ir = rsqrtf(r2); ca = bwc * ir; sa = aws * ir; }
        float c1 = A_OBS_C * A_OBS_C * ca * ca + B_OBS_C * B_OBS_C * sa * sa;
        float c2 = A_OBS_C * wc * ca + B_OBS_C * ws * sa;
        float dob = fmaxf(1.0f, c2 / c1);
        sux += A_OBS_C * dob * ca;
        suy += B_OBS_C * dob * sa;
      }
#pragma unroll
      for (int j = 0; j < NV; ++j) {
        float pj = sP[t * NV + j], pdj = sPd[t * NV + j], pddj = sPdd[t * NV + j];
        aUX[j] += sux * pj;   aUY[j] += suy * pj;
        aAX[j] += daca * pddj; aAY[j] += dasa * pddj;
        aVX[j] += dvcv * pdj;  aVY[j] += dvsv * pdj;
        aL1[j] += l1 * pj;     aL2[j] += l2 * pj;
      }
    }
#pragma unroll
    for (int j = 0; j < NV; ++j) {
      aUX[j] = wave_sum(aUX[j]); aUY[j] = wave_sum(aUY[j]);
      aAX[j] = wave_sum(aAX[j]); aAY[j] = wave_sum(aAY[j]);
      aVX[j] = wave_sum(aVX[j]); aVY[j] = wave_sum(aVY[j]);
      aL1[j] = wave_sum(aL1[j]); aL2[j] = wave_sum(aL2[j]);
    }
    float linx[NV], liny[NV];
#pragma unroll
    for (int k = 0; k < NV; ++k) {
      float ptpcx = 0.f, ptpcy = 0.f, pdcx = 0.f, pdcy = 0.f, pddcx = 0.f, pddcy = 0.f;
#pragma unroll
      for (int j = 0; j < NV; ++j) {
        float g0 = sPtP[k * NV + j], g1 = sPdTPd[k * NV + j], g2 = sPddTPdd[k * NV + j];
        ptpcx += g0 * cx[j]; ptpcy += g0 * cy[j];
        pdcx  += g1 * cx[j]; pdcy  += g1 * cy[j];
        pddcx += g2 * cx[j]; pddcy += g2 * cy[j];
      }
      // rxo@A = 10*PtP*cx - xoP - Su ; rax@Pdd = Pdd'Pdd*cx - S_daca ; rvx@Pd similar
      lx[k] -= RHO_OBS_C * (10.f * ptpcx - xoP[k] - aUX[k])
             + RHO_INEQ_C * (pddcx - aAX[k]) + RHO_INEQ_C * (pdcx - aVX[k]);
      ly[k] -= RHO_OBS_C * (10.f * ptpcy - yoP[k] - aUY[k])
             + RHO_INEQ_C * (pddcy - aAY[k]) + RHO_INEQ_C * (pdcy - aVY[k])
             + RHO_LANE_C * aL1[k];
      linx[k] = -lx[k] - RHO_PROJ_C * cxb[k] - RHO_OBS_C * (xoP[k] + aUX[k])
              - RHO_INEQ_C * aAX[k] - RHO_INEQ_C * aVX[k];
      liny[k] = -ly[k] - RHO_PROJ_C * cyb[k] - RHO_OBS_C * (yoP[k] + aUY[k])
              - RHO_INEQ_C * aAY[k] - RHO_INEQ_C * aVY[k] - RHO_LANE_C * aL2[k];
    }
    float ncx[NV], ncy[NV];
#pragma unroll
    for (int k = 0; k < NV; ++k) {
      float sx = sInv2x[k * 14 + 12] * beqx1;
      float sy = sInv2y[k * 15 + 12] * beqy1;
#pragma unroll
      for (int j = 0; j < NV; ++j) { sx -= sInv2x[k * 14 + j] * linx[j]; sy -= sInv2y[k * 15 + j] * liny[j]; }
      ncx[k] = sx; ncy[k] = sy;
    }
#pragma unroll
    for (int k = 0; k < NV; ++k) { cx[k] = ncx[k]; cy[k] = ncy[k]; }
  }

  if (lane == 0) {
#pragma unroll
    for (int k = 0; k < NV; ++k) {
      out[b * 22 + k] = cx[k];
      out[b * 22 + 11 + k] = cy[k];
    }
  }
}

// ---------------------------------------------------------------------------
extern "C" void kernel_launch(void* const* d_in, const int* in_sizes, int n_in,
                              void* d_out, int out_size, void* d_ws, size_t ws_size,
                              hipStream_t stream)
{
  (void)in_sizes; (void)n_in; (void)out_size; (void)ws_size;
  const float* inp  = (const float*)d_in[0];
  const float* ego  = (const float*)d_in[1];
  const float* traj = (const float*)d_in[2];
  const float* yub  = (const float*)d_in[3];
  const float* ylb  = (const float*)d_in[4];
  const float* eps  = (const float*)d_in[5];
  const float* ew1  = (const float*)d_in[6];
  const float* eb1  = (const float*)d_in[7];
  const float* ew2  = (const float*)d_in[8];
  const float* eb2  = (const float*)d_in[9];
  const float* ewmu = (const float*)d_in[10];
  const float* ebmu = (const float*)d_in[11];
  const float* ewlv = (const float*)d_in[12];
  const float* eblv = (const float*)d_in[13];
  const float* dw1  = (const float*)d_in[14];
  const float* db1  = (const float*)d_in[15];
  const float* dw2  = (const float*)d_in[16];
  const float* db2  = (const float*)d_in[17];
  const float* dw3  = (const float*)d_in[18];
  const float* db3  = (const float*)d_in[19];
  const float* P    = (const float*)d_in[20];
  const float* Pd   = (const float*)d_in[21];
  const float* Pdd  = (const float*)d_in[22];
  const float* mean = (const float*)d_in[23];
  const float* stdv = (const float*)d_in[24];
  float* out = (float*)d_out;

  size_t off = 0;
  auto take = [&](size_t bytes) -> void* {
    off = (off + 255) & ~(size_t)255;
    void* p = (char*)d_ws + off;
    off += bytes;
    return p;
  };
  float*    cpack = (float*)take(CPACK_N * sizeof(float));
  _Float16* x1    = (_Float16*)take((size_t)1024 * 256 * 2);
  _Float16* w1t   = (_Float16*)take((size_t)1024 * 256 * 2);
  _Float16* w2t   = (_Float16*)take((size_t)1024 * 1024 * 2);
  _Float16* h1    = (_Float16*)take((size_t)1024 * 1024 * 2);
  _Float16* h2    = (_Float16*)take((size_t)1024 * 1024 * 2);
  float*    zbuf  = (float*)take((size_t)1024 * 2 * 4);
  float*    inpn  = (float*)take((size_t)1024 * 55 * 4);
  _Float16* x2    = (_Float16*)take((size_t)1024 * 64 * 2);
  _Float16* dw1t  = (_Float16*)take((size_t)1024 * 64 * 2);
  _Float16* dw2t  = (_Float16*)take((size_t)1024 * 1024 * 2);
  _Float16* h3    = (_Float16*)take((size_t)1024 * 1024 * 2);
  _Float16* h4    = (_Float16*)take((size_t)1024 * 1024 * 2);
  float*    nnout = (float*)take((size_t)1024 * 8 * 4);

  // setup (independent)
  prep_const_kernel<<<1, 256, 0, stream>>>(P, Pd, Pdd, cpack);
  transpose_pad_f16<<<512, 256, 0, stream>>>(ew1, w1t, 255, 1024, 256);
  transpose_pad_f16<<<1024, 256, 0, stream>>>(ew2, w2t, 1024, 1024, 1024);
  transpose_pad_f16<<<256, 256, 0, stream>>>(dw1, dw1t, 57, 1024, 64);
  transpose_pad_f16<<<1024, 256, 0, stream>>>(dw2, dw2t, 1024, 1024, 1024);
  build_x1_kernel<<<512, 256, 0, stream>>>(inp, traj, mean, stdv, x1, inpn);

  // encoder: 32x64 macro-tiles -> (1024/32)*(1024/64)=512 waves = 64 blocks
  wmma_gemm_relu<<<64, 256, 0, stream>>>(x1, w1t, eb1, h1, 1024, 1024, 256, 1);
  wmma_gemm_relu<<<64, 256, 0, stream>>>(h1, w2t, eb2, h2, 1024, 1024, 1024, 1);
  muvarz_kernel<<<1024, 128, 0, stream>>>(h2, ewmu, ebmu, ewlv, eblv, eps, zbuf);

  // decoder
  build_x2_kernel<<<256, 256, 0, stream>>>(zbuf, inpn, x2);
  wmma_gemm_relu<<<64, 256, 0, stream>>>(x2, dw1t, db1, h3, 1024, 1024, 64, 1);
  wmma_gemm_relu<<<64, 256, 0, stream>>>(h3, dw2t, db2, h4, 1024, 1024, 1024, 1);
  dec3_kernel<<<1024, 256, 0, stream>>>(h4, dw3, db3, nnout);

  // projection solver: wave per row, 256 blocks x 4 waves = 1024 rows
  solver_kernel<<<256, 128, 0, stream>>>(inp, ego, yub, ylb, nnout, P, Pd, Pdd,
                                         cpack, out);
}
